// ChannelTransformer_38646115730064
// MI455X (gfx1250) — matrix operand order repaired and checked by reading, not compiled, not run on hardware
//
#include <hip/hip_runtime.h>
#include <hip/hip_bf16.h>

typedef __bf16 bf16_t;
typedef bf16_t v16bf __attribute__((ext_vector_type(16)));
typedef bf16_t v8bf  __attribute__((ext_vector_type(8)));
typedef float  v8f   __attribute__((ext_vector_type(8)));
typedef unsigned u32x4 __attribute__((ext_vector_type(4)));
typedef int      i32x4 __attribute__((ext_vector_type(4)));
typedef int      i32x8 __attribute__((ext_vector_type(8)));

#define DEVI static __device__ __forceinline__

// fp32 -> bf16 round-to-nearest-even
DEVI bf16_t f2bf(float f) {
  unsigned u = __builtin_bit_cast(unsigned, f);
  unsigned r = (u + 0x7fffu + ((u >> 16) & 1u)) >> 16;
  return __builtin_bit_cast(bf16_t, (unsigned short)r);
}

DEVI v16bf cat8(v8bf lo, v8bf hi) {
  return __builtin_shufflevector(lo, hi, 0,1,2,3,4,5,6,7,8,9,10,11,12,13,14,15);
}

// ---------------------------------------------------------------------------
// Generic batched GEMM:  C[z] = epilogue( alpha * A[z] * B[z]^T )
//   A: bf16 [M,K] row-major, B: bf16 [N,K] row-major (both K-contiguous)
// 128 threads = 4 waves (2x2), block tile 64x64, wave tile 32x32
// (2x2 v_wmma_f32_16x16x32_bf16). All GEMMs here have M%64==0, N%64==0, K%32==0.
// TDM=true: the block's whole B panel [64 x K] is staged into LDS by one
// Tensor Data Mover DMA (tensor_load_to_lds, TENSORcnt), then fragments are
// read with ds_load_b128 while A streams from global.
// ---------------------------------------------------------------------------
struct GemmArgs {
  const bf16_t* A; const bf16_t* B;
  long long sA, sB;          // batch strides (elements)
  int lda, ldb;              // row strides (elements)
  int M, N, K;
  float alpha;
  const float* bias;         // +bias[n]                         (optional)
  const float* affg; const float* affb; float affs;  // v = v*(g[n]*affs)+b[n] (optional)
  int act;                   // 0 none, 1 exact gelu, 2 relu
  const float* R;            // fp32 residual, added after activation (optional)
  long long r_off, r_sb; int r_msplit; long long r_mlo, r_mhi, r_n;
  float*  C;                 // fp32 out (exactly one of C / Cb set)
  bf16_t* Cb;                // bf16 out
  long long c_off, c_sb; int c_msplit; long long c_mlo, c_mhi, c_n;
};

DEVI void epilogue_tile(const GemmArgs& g, int z, int mB, int nB, v8f acc) {
  int lane = threadIdx.x & 31;
  int lr = lane & 15, hi = lane >> 4;
  int n = nB + lr;
#pragma unroll
  for (int r = 0; r < 8; r++) {
    int m = mB + r + (hi << 3);   // C/D layout: VGPR r -> row r (+8 for upper lane half)
    float v = g.alpha * acc[r];
    if (g.bias) v += g.bias[n];
    if (g.affg) v = v * (g.affg[n] * g.affs) + g.affb[n];
    if (g.act == 1)      v = 0.5f * v * (1.0f + erff(v * 0.70710678118654752f));
    else if (g.act == 2) v = fmaxf(v, 0.0f);
    if (g.R) {
      long long ri = g.r_off + (long long)z * g.r_sb
                   + (long long)(m % g.r_msplit) * g.r_mlo
                   + (long long)(m / g.r_msplit) * g.r_mhi
                   + (long long)n * g.r_n;
      v += g.R[ri];
    }
    long long ci = g.c_off + (long long)z * g.c_sb
                 + (long long)(m % g.c_msplit) * g.c_mlo
                 + (long long)(m / g.c_msplit) * g.c_mhi
                 + (long long)n * g.c_n;
    if (g.C) g.C[ci] = v; else g.Cb[ci] = f2bf(v);
  }
}

template <bool TDM>
__global__ __launch_bounds__(128) void k_gemm(GemmArgs g) {
  extern __shared__ bf16_t smem[];   // TDM: B panel [64][K], rows packed
  const int z  = blockIdx.z;
  const int m0 = blockIdx.y << 6, n0 = blockIdx.x << 6;
  const int wave  = threadIdx.x >> 5;
  const int mBase = m0 + ((wave >> 1) << 5);
  const int nLoc  = (wave & 1) << 5;            // local col base within 64-wide panel
  const int lane = threadIdx.x & 31;
  const int lr = lane & 15, hi = lane >> 4;

  if constexpr (TDM) {
    // One DMA per block: B panel [64 rows x K] -> LDS. D# per ISA 8.3/8.4.
    if (threadIdx.x < 32) {
      unsigned long long ga =
          (unsigned long long)(size_t)(const void*)(g.B + (long long)z * g.sB
                                                        + (long long)n0 * g.ldb);
      unsigned ldsOff = (unsigned)(size_t)(void*)smem;   // addr[31:0] = LDS offset
      unsigned kk = (unsigned)g.K;
      u32x4 G0 = { 1u,                                   // count=1, user descriptor
                   ldsOff,                               // lds_addr (bytes)
                   (unsigned)ga,                         // global_addr[31:0]
                   ((unsigned)(ga >> 32) & 0x01ffffffu) | 0x80000000u };  // [56:32] | type=2
      i32x8 G1 = { (int)0x10000u,                        // data_size=1 (2-byte)
                   (int)((kk & 0xffffu) << 16),          // tensor_dim0[15:0]
                   (int)((kk >> 16) | (64u << 16)),      // tensor_dim0[31:16] | tensor_dim1=64
                   (int)((kk & 0xffffu) << 16),          // tile_dim0 = K
                   (int)64u,                             // tile_dim1 = 64, tile_dim2 = 0
                   (int)(unsigned)g.ldb,                 // tensor_dim0_stride[31:0]
                   0, 0 };
      i32x4 G2 = {}; i32x4 G3 = {};
#if defined(__clang_major__) && __clang_major__ >= 23
      i32x8 G4 = {};
      __builtin_amdgcn_tensor_load_to_lds(G0, G1, G2, G3, G4, 0);
#else
      __builtin_amdgcn_tensor_load_to_lds(G0, G1, G2, G3, 0);
#endif
      __builtin_amdgcn_s_wait_tensorcnt(0);
    }
    __syncthreads();
  }

  // A fragment: lane<16 -> row=lane, K 0..15;
  //             lane>=16 -> row=lane-16, K 16..31
  const bf16_t* pa0 = g.A + (long long)z * g.sA + (long long)(mBase + lr) * g.lda + (hi << 4);
  const bf16_t* pa1 = pa0 + 16LL * g.lda;
  // B fragment: lane<16 -> col=lane, K 0..15; lane>=16 -> col=lane-16, K 16..31
  const bf16_t* pb0;
  const bf16_t* pb1;
  if constexpr (TDM) {
    pb0 = smem + (size_t)(nLoc + lr) * (size_t)g.K + (hi << 4);
    pb1 = pb0 + (size_t)16 * g.K;
  } else {
    pb0 = g.B + (long long)z * g.sB + (long long)(n0 + nLoc + lr) * g.ldb + (hi << 4);
    pb1 = pb0 + 16LL * g.ldb;
  }

  v8f acc00 = {}, acc01 = {}, acc10 = {}, acc11 = {};
  for (int k0 = 0; k0 < g.K; k0 += 32) {
    v16bf a0 = cat8(*(const v8bf*)(pa0 + k0), *(const v8bf*)(pa0 + k0 + 8));
    v16bf a1 = cat8(*(const v8bf*)(pa1 + k0), *(const v8bf*)(pa1 + k0 + 8));
    v16bf b0 = cat8(*(const v8bf*)(pb0 + k0), *(const v8bf*)(pb0 + k0 + 8));
    v16bf b1 = cat8(*(const v8bf*)(pb1 + k0), *(const v8bf*)(pb1 + k0 + 8));
    if (k0 + 32 < g.K) {            // cover next-tile latency (global_prefetch_b8)
      __builtin_prefetch(pa0 + k0 + 32, 0, 3);
      __builtin_prefetch(pa1 + k0 + 32, 0, 3);
      if constexpr (!TDM) {
        __builtin_prefetch(pb0 + k0 + 32, 0, 3);
        __builtin_prefetch(pb1 + k0 + 32, 0, 3);
      }
    }
    acc00 = __builtin_amdgcn_wmma_f32_16x16x32_bf16(false, a0, false, b0, (short)0, acc00, false, false);
    acc01 = __builtin_amdgcn_wmma_f32_16x16x32_bf16(false, a0, false, b1, (short)0, acc01, false, false);
    acc10 = __builtin_amdgcn_wmma_f32_16x16x32_bf16(false, a1, false, b0, (short)0, acc10, false, false);
    acc11 = __builtin_amdgcn_wmma_f32_16x16x32_bf16(false, a1, false, b1, (short)0, acc11, false, false);
  }
  const int nBase = n0 + nLoc;
  epilogue_tile(g, z, mBase,      nBase,      acc00);
  epilogue_tile(g, z, mBase,      nBase + 16, acc01);
  epilogue_tile(g, z, mBase + 16, nBase,      acc10);
  epilogue_tile(g, z, mBase + 16, nBase + 16, acc11);
}

// ---------------------------------------------------------------------------
// Elementwise / reduction kernels (wave32 shfl + LDS block reductions)
// ---------------------------------------------------------------------------
DEVI float block_reduce(float v, float* sh, bool ismax) {
#pragma unroll
  for (int o = 16; o > 0; o >>= 1) {
    float t = __shfl_xor(v, o, 32);
    v = ismax ? fmaxf(v, t) : (v + t);
  }
  int w = threadIdx.x >> 5, nw = blockDim.x >> 5;
  if ((threadIdx.x & 31) == 0) sh[w] = v;
  __syncthreads();
  if (threadIdx.x < 32) {
    float t2 = (threadIdx.x < (unsigned)nw) ? sh[threadIdx.x] : (ismax ? -3.0e38f : 0.0f);
#pragma unroll
    for (int o = 16; o > 0; o >>= 1) {
      float t = __shfl_xor(t2, o, 32);
      t2 = ismax ? fmaxf(t2, t) : (t2 + t);
    }
    if (threadIdx.x == 0) sh[0] = t2;
  }
  __syncthreads();
  float r = sh[0];
  __syncthreads();
  return r;
}

__global__ void k_cvt(const float* s, bf16_t* d, long long n) {
  long long i = (long long)blockIdx.x * blockDim.x + threadIdx.x;
  if (i < n) d[i] = f2bf(s[i]);
}

__global__ __launch_bounds__(256) void k_layernorm(const float* x, int ld, long long off, int C,
                                                   const float* gw, const float* bw,
                                                   bf16_t* out, int ldo) {
  __shared__ float sh[32];
  long long row = blockIdx.x;
  const float* xr = x + off + row * ld;
  float s = 0.f, s2 = 0.f;
  for (int c = threadIdx.x; c < C; c += blockDim.x) { float t = xr[c]; s += t; s2 += t * t; }
  s  = block_reduce(s,  sh, false);
  s2 = block_reduce(s2, sh, false);
  float mu  = s / (float)C;
  float var = s2 / (float)C - mu * mu;
  float isd = rsqrtf(var + 1e-6f);
  bf16_t* orow = out + row * (long long)ldo;
  for (int c = threadIdx.x; c < C; c += blockDim.x)
    orow[c] = f2bf((xr[c] - mu) * isd * gw[c] + bw[c]);
}

__global__ __launch_bounds__(256) void k_instnorm_stats(const float* s, long long n, float* stats) {
  __shared__ float sh[32];
  long long z = blockIdx.x;
  const float* p = s + z * n;
  float a = 0.f, b = 0.f;
  for (long long i = threadIdx.x; i < n; i += blockDim.x) { float t = p[i]; a += t; b += t * t; }
  a = block_reduce(a, sh, false);
  b = block_reduce(b, sh, false);
  if (threadIdx.x == 0) {
    float mu  = a / (float)n;
    float var = b / (float)n - mu * mu;
    stats[z * 2]     = mu;
    stats[z * 2 + 1] = rsqrtf(var + 1e-5f);
  }
}

// instance-norm + softmax; row cached in registers (N=960 <= 4*256); bf16 probs out
__global__ __launch_bounds__(256) void k_softmax(const float* s, bf16_t* pb, const float* stats,
                                                 int C, int N) {
  __shared__ float sh[32];
  int z = blockIdx.y, c = blockIdx.x;
  float mu = stats[z * 2], isd = stats[z * 2 + 1];
  const float* row  = s  + ((long long)z * C + c) * N;
  bf16_t*      prow = pb + ((long long)z * C + c) * N;
  float vv[4];
  int nj = 0;
  float mx = -3.0e38f;
  for (int k = threadIdx.x; k < N; k += blockDim.x) {
    float t = (row[k] - mu) * isd;
    vv[nj++] = t; mx = fmaxf(mx, t);
  }
  mx = block_reduce(mx, sh, true);
  float sum = 0.f;
  for (int j = 0; j < nj; j++) { float e = expf(vv[j] - mx); vv[j] = e; sum += e; }
  sum = block_reduce(sum, sh, false);
  float inv = 1.0f / sum;
  int j = 0;
  for (int k = threadIdx.x; k < N; k += blockDim.x) prow[k] = f2bf(vv[j++] * inv);
}

__global__ void k_reduce_heads(const float* ch, bf16_t* ctx, int C, long long total) {
  long long i = (long long)blockIdx.x * blockDim.x + threadIdx.x;
  if (i >= total) return;
  long long m = i / C; int c = (int)(i % C);
  long long b = m >> 6, p = m & 63;
  float s = 0.f;
#pragma unroll
  for (int h = 0; h < 4; h++) s += ch[(((b * 4 + h) * 64) + p) * C + c];
  ctx[i] = f2bf(0.25f * s);
}

__global__ void k_patchify(const float* en, bf16_t* out, int C, int p, int F, long long total) {
  long long i = (long long)blockIdx.x * blockDim.x + threadIdx.x;
  if (i >= total) return;
  int Kc = C * p * p;
  long long m = i / Kc; int k = (int)(i % Kc);
  int b = (int)(m >> 6), patch = (int)(m & 63);
  int ph = patch >> 3, pw = patch & 7;
  int c = k / (p * p), r = k % (p * p);
  int kh = r / p, kw = r % p;
  out[i] = f2bf(en[(((long long)b * C + c) * F + ph * p + kh) * F + pw * p + kw]);
}

__global__ void k_expand(const bf16_t* xln, bf16_t* out, int C, int p, int F, long long total) {
  long long i = (long long)blockIdx.x * blockDim.x + threadIdx.x;
  if (i >= total) return;
  long long m = i / C; int c = (int)(i % C);
  int b = (int)(m / (F * F)); int rem = (int)(m % (F * F));
  int h = rem / F, w = rem % F;
  int patch = (h / p) * 8 + (w / p);
  out[i] = xln[((long long)b * 64 + patch) * C + c];
}

// ---------------------------------------------------------------------------
// Host orchestration
// ---------------------------------------------------------------------------
extern "C" void kernel_launch(void* const* d_in, const int* in_sizes, int n_in,
                              void* d_out, int out_size, void* d_ws, size_t ws_size,
                              hipStream_t stream) {
  (void)in_sizes; (void)n_in; (void)out_size;
  static const int CD[4]  = {64, 128, 256, 512};
  static const int OFF[4] = {0, 64, 192, 448};
  static const int PSS[4] = {8, 4, 2, 1};
  static const int FMS[4] = {64, 32, 16, 8};
  const int KV = 960;
  const long long OUTOFF[4] = {0, 1048576, 1572864, 1835008};

  auto IN = [&](int idx) { return (const float*)d_in[idx]; };
  const float* en[4]; for (int i = 0; i < 4; i++) en[i] = IN(i);

  // ---- workspace carve-out (1KB aligned regions) ----
  size_t off = 0;
  auto allocB = [&](size_t bytes) {
    char* p = (char*)d_ws + off;
    off = (off + bytes + 1023) & ~(size_t)1023;
    return p;
  };
  float*  emb = (float*) allocB(245760ull * 4);   // residual state [B,P,KV] fp32
  bf16_t* eln = (bf16_t*)allocB(245760ull * 2);   // LN(concat)     [B,P,KV]
  bf16_t* cx  = (bf16_t*)allocB(131072ull * 2);   // per-scale LN   [B*P,<=512]
  bf16_t* kt  = (bf16_t*)allocB(983040ull * 2);   // K^T [B,H,KV,P]
  bf16_t* vb  = (bf16_t*)allocB(983040ull * 2);   // V   [B,H,P,KV]
  bf16_t* qt  = (bf16_t*)allocB(524288ull * 2);   // Q^T [B,H,C,P]
  float*  sb  = (float*) allocB(7864320ull * 4);  // scores fp32 [B*H,C,KV]
  bf16_t* pb  = (bf16_t*)allocB(7864320ull * 2);  // probs bf16
  float*  chh = (float*) allocB(524288ull * 4);   // ctx per head fp32
  bf16_t* ctx = (bf16_t*)allocB(131072ull * 2);   // ctx [B*P,C]
  bf16_t* h1  = (bf16_t*)allocB(524288ull * 2);   // MLP hidden [B*P,4C]
  bf16_t* ar  = (bf16_t*)allocB(1048576ull * 2);  // patch / upsample staging
  float*  st  = (float*) allocB(128);             // instnorm stats [16][2]

  // ---- bf16 weight arena: convert once per launch (L2-resident thereafter) ----
  bf16_t* wEmb[4]; bf16_t* wRec[4]; bf16_t* wK[4]; bf16_t* wV[4];
  bf16_t* wQ[4][4]; bf16_t* wO[4][4]; bf16_t* wF1[4][4]; bf16_t* wF2[4][4];
  struct Cvt { const float* s; bf16_t* d; long long n; };
  Cvt cv[96]; int ncv = 0;
  auto wconv = [&](int idx, long long n) {
    bf16_t* d = (bf16_t*)allocB((size_t)n * 2);
    cv[ncv++] = Cvt{ IN(idx), d, n };
    return d;
  };
  for (int i = 0; i < 4; i++)
    wEmb[i] = wconv(4 + 3 * i, (long long)CD[i] * CD[i] * PSS[i] * PSS[i]);
  for (int l = 0; l < 4; l++) {
    int L0 = 16 + 44 * l;
    wK[l] = wconv(L0 + 2, 4LL * KV * KV);
    wV[l] = wconv(L0 + 3, 4LL * KV * KV);
    for (int i = 0; i < 4; i++) {
      int S = L0 + 4 + 10 * i; long long C = CD[i];
      wQ[l][i]  = wconv(S + 2, 4 * C * C);
      wO[l][i]  = wconv(S + 3, C * C);
      wF1[l][i] = wconv(S + 6, 4 * C * C);
      wF2[l][i] = wconv(S + 8, 4 * C * C);
    }
  }
  for (int i = 0; i < 4; i++)
    wRec[i] = wconv(192 + 6 * i + 2, (long long)CD[i] * CD[i]);
  if (off > ws_size) return;

  auto ew = [&](long long total) { return dim3((unsigned)((total + 255) / 256)); };
  for (int j = 0; j < ncv; j++)
    k_cvt<<<ew(cv[j].n), 256, 0, stream>>>(cv[j].s, cv[j].d, cv[j].n);

  auto mkgemm = [](const bf16_t* A, int lda, long long sA,
                   const bf16_t* Bt, int ldb, long long sB,
                   int M, int N, int K, float* C, bf16_t* Cb) {
    GemmArgs g{}; g.A = A; g.lda = lda; g.sA = sA; g.B = Bt; g.ldb = ldb; g.sB = sB;
    g.M = M; g.N = N; g.K = K; g.alpha = 1.0f; g.affs = 1.0f;
    g.c_msplit = 1 << 30; g.c_mlo = N; g.c_mhi = 0; g.c_n = 1;   // default dense [M,N]
    g.r_msplit = 1 << 30; g.C = C; g.Cb = Cb;
    return g;
  };
  auto rungemm = [&](const GemmArgs& g, int Z) {
    dim3 grid((unsigned)(g.N / 64), (unsigned)(g.M / 64), (unsigned)Z);
    if (g.K <= 1024) {   // B panel [64 x K] bf16 fits 128KB LDS -> TDM staging
      size_t lds = (size_t)g.K * 64 * 2;
      k_gemm<true><<<grid, 128, lds, stream>>>(g);
    } else {
      k_gemm<false><<<grid, 128, 0, stream>>>(g);
    }
  };

  // ---------------- Patch embedding (per scale) -> emb_all slices ----------
  for (int i = 0; i < 4; i++) {
    int C = CD[i], p = PSS[i], F = FMS[i], Kc = C * p * p;
    long long tot = 256LL * Kc;
    k_patchify<<<ew(tot), 256, 0, stream>>>(en[i], ar, C, p, F, tot);
    GemmArgs g = mkgemm(ar, Kc, 0, wEmb[i], Kc, 0, 256, C, Kc, emb, nullptr);
    g.bias = IN(5 + 3 * i);
    g.R = IN(6 + 3 * i);                    // positional embedding [P, C]
    g.r_msplit = 64; g.r_mlo = C; g.r_mhi = 0; g.r_n = 1;
    g.c_mlo = KV; g.c_off = OFF[i];
    rungemm(g, 1);
  }

  // ---------------- Encoder layers -----------------------------------------
  for (int l = 0; l < 4; l++) {
    int L0 = 16 + 44 * l;

    k_layernorm<<<256, 256, 0, stream>>>(emb, KV, 0, KV, IN(L0), IN(L0 + 1), eln, KV);

    // K^T[b,h,k,p]
    {
      GemmArgs g = mkgemm(eln, KV, 0, wK[l], KV, (long long)KV * KV, 256, KV, KV, nullptr, kt);
      g.c_msplit = 64; g.c_mlo = 1; g.c_mhi = 4LL * KV * 64; g.c_n = 64; g.c_sb = (long long)KV * 64;
      rungemm(g, 4);
    }
    // V[b,h,p,k]
    {
      GemmArgs g = mkgemm(eln, KV, 0, wV[l], KV, (long long)KV * KV, 256, KV, KV, nullptr, vb);
      g.c_msplit = 64; g.c_mlo = KV; g.c_mhi = 4LL * 64 * KV; g.c_n = 1; g.c_sb = 64LL * KV;
      rungemm(g, 4);
    }

    for (int i = 0; i < 4; i++) {
      int C = CD[i], off_c = OFF[i], S = L0 + 4 + 10 * i;

      k_layernorm<<<256, 256, 0, stream>>>(emb, KV, off_c, C, IN(S), IN(S + 1), cx, C);

      // Q^T[b,h,d,p]
      {
        GemmArgs g = mkgemm(cx, C, 0, wQ[l][i], C, (long long)C * C, 256, C, C, nullptr, qt);
        g.c_msplit = 64; g.c_mlo = 1; g.c_mhi = 4LL * C * 64; g.c_n = 64; g.c_sb = (long long)C * 64;
        rungemm(g, 4);
      }
      // scores s[z=(b,h)][c][k] = (Q^T K) / sqrt(KV)   (fp32)
      {
        GemmArgs g = mkgemm(qt, 64, (long long)C * 64, kt, 64, (long long)KV * 64, C, KV, 64, sb, nullptr);
        g.alpha = 0.032274861218395144f;    // 1/sqrt(960)
        g.c_sb = (long long)C * KV;
        rungemm(g, 16);
      }
      k_instnorm_stats<<<16, 256, 0, stream>>>(sb, (long long)C * KV, st);
      k_softmax<<<dim3((unsigned)C, 16), 256, 0, stream>>>(sb, pb, st, C, KV);

      // ctx_h[z][p][c] = V probs^T   (fp32)
      {
        GemmArgs g = mkgemm(vb, KV, 64LL * KV, pb, KV, (long long)C * KV, 64, C, KV, chh, nullptr);
        g.c_sb = 64LL * C;
        rungemm(g, 16);
      }
      k_reduce_heads<<<ew(256LL * C), 256, 0, stream>>>(chh, ctx, C, 256LL * C);

      // out-proj + residual, in-place into emb_all slice
      {
        GemmArgs g = mkgemm(ctx, C, 0, wO[l][i], C, 0, 256, C, C, emb, nullptr);
        g.R = emb; g.r_mlo = KV; g.r_n = 1; g.r_off = off_c;
        g.c_mlo = KV; g.c_off = off_c;
        rungemm(g, 1);
      }
      // MLP
      k_layernorm<<<256, 256, 0, stream>>>(emb, KV, off_c, C, IN(S + 4), IN(S + 5), cx, C);
      {
        GemmArgs g = mkgemm(cx, C, 0, wF1[l][i], C, 0, 256, 4 * C, C, nullptr, h1);
        g.bias = IN(S + 7); g.act = 1;      // exact GELU
        rungemm(g, 1);
      }
      {
        GemmArgs g = mkgemm(h1, 4 * C, 0, wF2[l][i], 4 * C, 0, 256, C, 4 * C, emb, nullptr);
        g.bias = IN(S + 9);
        g.R = emb; g.r_mlo = KV; g.r_n = 1; g.r_off = off_c;
        g.c_mlo = KV; g.c_off = off_c;
        rungemm(g, 1);
      }
    }
  }

  // ---------------- Final LN + reconstruct + input residual ----------------
  for (int i = 0; i < 4; i++) {
    int C = CD[i], p = PSS[i], F = FMS[i], F0 = 192 + 6 * i;
    k_layernorm<<<256, 256, 0, stream>>>(emb, KV, OFF[i], C, IN(F0), IN(F0 + 1), cx, C);
    long long tot = 4LL * F * F * C;
    k_expand<<<ew(tot), 256, 0, stream>>>(cx, ar, C, p, F, tot);
    GemmArgs g = mkgemm(ar, C, 0, wRec[i], C, 0, 4 * F * F, C, C, (float*)d_out, nullptr);
    g.bias = IN(F0 + 3);
    g.affg = IN(F0 + 4); g.affb = IN(F0 + 5);
    g.affs = 0.99999500003749969f;          // 1/sqrt(1+1e-5): BN eval, unit running stats
    g.act = 2;                               // ReLU
    g.R = en[i];                             // + input residual
    g.r_msplit = F * F; g.r_mlo = 1; g.r_mhi = (long long)C * F * F; g.r_n = (long long)F * F;
    g.c_msplit = F * F; g.c_mlo = 1; g.c_mhi = (long long)C * F * F; g.c_n = (long long)F * F;
    g.c_off = OUTOFF[i];
    rungemm(g, 1);
  }
}
